// StatefulQwen3Decoder_25890062861086
// MI455X (gfx1250) — matrix operand order repaired and checked by reading, not compiled, not run on hardware
//
#include <hip/hip_runtime.h>
#include <hip/hip_bf16.h>

typedef __attribute__((ext_vector_type(16))) _Float16 v16h;
typedef __attribute__((ext_vector_type(8)))  float    v8f;

#define SEQ 512
#define DMODEL 1024
#define NHQ 16
#define NHKV 8
#define HD 128
#define DFF 3072
#define NLAYER 28

// column stride (in halfs) of the LDS-staged B tile: padded 32->36 to break
// power-of-2 bank collisions while keeping every column 8-byte aligned.
#define BT_STRIDE 36

// ---------------------------------------------------------------------------
// WMMA fragment helpers, wave32; A and B both hold k = 16*(lane/16) + s (contiguous order).
// ---------------------------------------------------------------------------

// A matrix 16x32 f16, from row-major [*, lda] f16.
// lane l<16: M=l, halves 0..15 = K d0..d0+15
// lane l>=16: M=l-16, halves 0..15 = K d0+16..d0+31
__device__ __forceinline__ v16h load_a_frag(const _Float16* A, int lda, int row0, int d0) {
    int lane = threadIdx.x & 31;
    const _Float16* p = A + (size_t)(row0 + (lane & 15)) * lda + d0 + ((lane >> 4) << 4);
    v16h a;
#pragma unroll
    for (int i = 0; i < 8; ++i) { a[i] = p[i]; a[i + 8] = p[8 + i]; }
    return a;
}

// B fragment from the LDS-staged, column-major tile: bt[col*BT_STRIDE + kk].
// lane l<16 holds n=nsub+l, K ks..ks+15 contiguous in LDS -> wide ds loads.
__device__ __forceinline__ v16h load_b_frag_lds(const _Float16* bt, int nsub) {
    int lane = threadIdx.x & 31;
    const _Float16* p = bt + (nsub + (lane & 15)) * BT_STRIDE + ((lane >> 4) << 4);
    v16h b;
#pragma unroll
    for (int i = 0; i < 16; ++i) b[i] = p[i];
    return b;
}

// B fragment when the source is stored K-major (i.e. the matmul's K dimension
// is the fastest-varying axis in memory): per lane 16 contiguous halfs.
// Used for both Q·K^T (Kmat row-major [key][d], K-dim = d) and
// P·V with V stored transposed [d][key] (K-dim = key).
__device__ __forceinline__ v16h load_b_frag_kmajor(const _Float16* B, int ldb, int n0, int k0) {
    int lane = threadIdx.x & 31;
    const _Float16* p = B + (size_t)(n0 + (lane & 15)) * ldb + k0 + ((lane >> 4) << 4);
    v16h b;
#pragma unroll
    for (int i = 0; i < 16; ++i) b[i] = p[i];
    return b;
}

__device__ __forceinline__ v8f wmma_f16(v16h a, v16h b, v8f c) {
    return __builtin_amdgcn_wmma_f32_16x16x32_f16(false, a, false, b, (short)0, c, false, false);
}

__device__ __forceinline__ float rowred_max(float v) {
#pragma unroll
    for (int m = 1; m < 16; m <<= 1) v = fmaxf(v, __shfl_xor(v, m, 32));
    return v;
}
__device__ __forceinline__ float rowred_sum(float v) {
#pragma unroll
    for (int m = 1; m < 16; m <<= 1) v += __shfl_xor(v, m, 32);
    return v;
}

// ---------------------------------------------------------------------------
// RMSNorm: x_f16[row, :] = h[row,:] * rsqrt(mean(h^2)+eps) * w
// ---------------------------------------------------------------------------
__global__ void __launch_bounds__(256) rmsnorm_f16_kernel(const float* __restrict__ h,
                                                          const float* __restrict__ w,
                                                          _Float16* __restrict__ xo) {
    int row = blockIdx.x;
    int tid = threadIdx.x;
    __shared__ float red[8];
    const float* hr = h + (size_t)row * DMODEL;
    float vals[4];
    float ss = 0.f;
#pragma unroll
    for (int i = 0; i < 4; ++i) { vals[i] = hr[tid + i * 256]; ss += vals[i] * vals[i]; }
#pragma unroll
    for (int m = 1; m < 32; m <<= 1) ss += __shfl_xor(ss, m, 32);
    if ((tid & 31) == 0) red[tid >> 5] = ss;
    __syncthreads();
    float tot = 0.f;
#pragma unroll
    for (int i = 0; i < 8; ++i) tot += red[i];
    float r = rsqrtf(tot * (1.0f / DMODEL) + 1e-6f);
#pragma unroll
    for (int i = 0; i < 4; ++i) {
        int idx = tid + i * 256;
        xo[(size_t)row * DMODEL + idx] = (_Float16)(vals[i] * r * w[idx]);
    }
}

// ---------------------------------------------------------------------------
// WMMA GEMM: C[M,N](f32) = A[M,K](f16,row-major) x W[K,N](f32,row-major).
// Block = 8 waves, tile 128M x 64N. Per 32-K step the whole block stages the
// 32x64 fp32 W slab through LDS (coalesced b128 row loads, one fp32->f16
// conversion total, column-major f16 layout) so each wave's 4 B fragments are
// contiguous wide ds loads. 4 WMMAs per wave per K-step share one A fragment.
// ---------------------------------------------------------------------------
template <bool RESID>
__global__ void __launch_bounds__(256) gemm_f16w32_kernel(const _Float16* __restrict__ A,
                                                          const float* __restrict__ W,
                                                          float* __restrict__ C,
                                                          const float* __restrict__ resid,
                                                          int M, int N, int K) {
    __shared__ __align__(16) _Float16 bt[64 * BT_STRIDE];

    const int tid   = threadIdx.y * 32 + threadIdx.x;
    const int nb    = blockIdx.x * 64;                       // block N base
    const int m0    = blockIdx.y * 128 + threadIdx.y * 16;   // wave M base
    // staging decomposition: thread -> (k-pair, 4 columns)
    const int kpair = tid >> 4;             // 0..15  -> kk = 2*kpair, 2*kpair+1
    const int ncol4 = (tid & 15) * 4;       // 0..60

    v8f acc[4];
#pragma unroll
    for (int s = 0; s < 4; ++s)
#pragma unroll
        for (int i = 0; i < 8; ++i) acc[s][i] = 0.f;

    for (int k0 = 0; k0 < K; k0 += 32) {
        // ---- stage W[k0..k0+31, nb..nb+63] -> LDS f16 column-major ----
        const float* r0 = W + (size_t)(k0 + 2 * kpair)     * N + nb + ncol4;
        const float* r1 = W + (size_t)(k0 + 2 * kpair + 1) * N + nb + ncol4;
        float a0[4], a1[4];
#pragma unroll
        for (int j = 0; j < 4; ++j) { a0[j] = r0[j]; a1[j] = r1[j]; }   // 2x b128
        if (k0 + 32 < K)
            __builtin_prefetch(W + (size_t)(k0 + 32 + 2 * kpair) * N + nb + ncol4, 0, 1);
        __syncthreads();   // previous step's fragment reads done
#pragma unroll
        for (int j = 0; j < 4; ++j) {
            _Float16* q = &bt[(ncol4 + j) * BT_STRIDE + 2 * kpair];
            q[0] = (_Float16)a0[j];
            q[1] = (_Float16)a1[j];
        }
        __syncthreads();   // tile visible to all waves

        // ---- 4 WMMAs: one A fragment, 4 N-subtiles from LDS ----
        v16h a = load_a_frag(A, K, m0, k0);
#pragma unroll
        for (int s = 0; s < 4; ++s) {
            v16h b = load_b_frag_lds(bt, s * 16);
            acc[s] = wmma_f16(a, b, acc[s]);
        }
    }

    const int lane = threadIdx.x & 31;
    const int mb   = m0 + ((lane >> 4) << 3);
#pragma unroll
    for (int s = 0; s < 4; ++s) {
        const int n = nb + s * 16 + (lane & 15);
#pragma unroll
        for (int r = 0; r < 8; ++r) {
            float val = acc[s][r];
            if (RESID) val += resid[(size_t)(mb + r) * N + n];
            C[(size_t)(mb + r) * N + n] = val;
        }
    }
}

// ---------------------------------------------------------------------------
// Per-head q/k RMSNorm + RoPE, plus v -> f16. One block per sequence row.
// q, k are written per-head row-major [H][SEQ][HD] (K-major for Q·K^T).
// v is written TRANSPOSED [HKV][HD][SEQ] so the P·V B fragment is a
// contiguous-per-lane read (key index fastest in memory).
// ---------------------------------------------------------------------------
__global__ void __launch_bounds__(256) qk_rope_kernel(const float* __restrict__ qf,
                                                      const float* __restrict__ kf,
                                                      const float* __restrict__ vf,
                                                      const float* __restrict__ cosb,
                                                      const float* __restrict__ sinb,
                                                      const float* __restrict__ qw,
                                                      const float* __restrict__ kw,
                                                      _Float16* __restrict__ qo,
                                                      _Float16* __restrict__ ko,
                                                      _Float16* __restrict__ voT) {
    int row  = blockIdx.x;
    int tid  = threadIdx.x;
    int lane = tid & 31;
    int wid  = tid >> 5;

    // v: cast to f16, store transposed [hh][d][row]
    for (int idx = tid; idx < NHKV * HD; idx += 256) {
        int hh = idx >> 7, d = idx & 127;
        voT[((size_t)hh * HD + d) * SEQ + row] = (_Float16)vf[(size_t)row * (NHKV * HD) + idx];
    }

    const float* cr = cosb + (size_t)row * HD;
    const float* sr = sinb + (size_t)row * HD;
    float c0 = cr[lane], c1 = cr[lane + 32], c2 = cr[lane + 64], c3 = cr[lane + 96];
    float s0 = sr[lane], s1 = sr[lane + 32], s2 = sr[lane + 64], s3 = sr[lane + 96];

    for (int g = wid; g < NHQ + NHKV; g += 8) {
        const float* src; const float* w; _Float16* dst;
        if (g < NHQ) {
            src = qf + (size_t)row * (NHQ * HD) + g * HD;  w = qw;
            dst = qo + ((size_t)g * SEQ + row) * HD;
        } else {
            int hh = g - NHQ;
            src = kf + (size_t)row * (NHKV * HD) + hh * HD; w = kw;
            dst = ko + ((size_t)hh * SEQ + row) * HD;
        }
        float v0 = src[lane], v1 = src[lane + 32], v2 = src[lane + 64], v3 = src[lane + 96];
        float ss = v0 * v0 + v1 * v1 + v2 * v2 + v3 * v3;
#pragma unroll
        for (int m = 1; m < 32; m <<= 1) ss += __shfl_xor(ss, m, 32);
        float r = rsqrtf(ss * (1.0f / HD) + 1e-6f);
        float n0 = v0 * r * w[lane],      n1 = v1 * r * w[lane + 32];
        float n2 = v2 * r * w[lane + 64], n3 = v3 * r * w[lane + 96];
        // rotate_half: d<64 -> -x[d+64]; d>=64 -> x[d-64]; +64 == +2 lane slots
        dst[lane]      = (_Float16)(n0 * c0 - n2 * s0);
        dst[lane + 32] = (_Float16)(n1 * c1 - n3 * s1);
        dst[lane + 64] = (_Float16)(n2 * c2 + n0 * s2);
        dst[lane + 96] = (_Float16)(n3 * c3 + n1 * s3);
    }
}

// ---------------------------------------------------------------------------
// Flash attention, causal, GQA. One wave per (q-head, 16-row q-tile).
// Q·K^T: B from K rows (d fastest).  P·V: B from V^T (key fastest).
// Both B fragments are contiguous-per-lane -> wide global loads.
// ---------------------------------------------------------------------------
__global__ void __launch_bounds__(32) flash_attn_kernel(const _Float16* __restrict__ Qm,
                                                        const _Float16* __restrict__ Km,
                                                        const _Float16* __restrict__ VmT,
                                                        _Float16* __restrict__ Om) {
    __shared__ _Float16 pls[16 * 32];
    const int h    = blockIdx.x;
    const int q0   = blockIdx.y * 16;
    const int kvh  = h >> 1;
    const int lane = threadIdx.x & 31;
    const int hi8  = (lane >> 4) << 3;
    const int ncol = lane & 15;
    const float scale = 0.08838834764831845f;   // 1/sqrt(128)

    const _Float16* Qh  = Qm  + (size_t)h   * SEQ * HD;
    const _Float16* Kh  = Km  + (size_t)kvh * SEQ * HD;   // [key][d]
    const _Float16* VhT = VmT + (size_t)kvh * HD * SEQ;   // [d][key]

    v16h qa[4];
#pragma unroll
    for (int c = 0; c < 4; ++c) qa[c] = load_a_frag(Qh, HD, q0, c * 32);

    v8f o[8];
    float rmax[8], rsum[8];
#pragma unroll
    for (int c = 0; c < 8; ++c) {
#pragma unroll
        for (int r = 0; r < 8; ++r) o[c][r] = 0.f;
        rmax[c] = -1e30f; rsum[c] = 0.f;
    }

    const int qend = q0 + 15;
    for (int k0 = 0; k0 <= qend; k0 += 32) {
        float p[2][8];
#pragma unroll
        for (int t = 0; t < 2; ++t) {
            int kk0 = k0 + t * 16;
            v8f s;
#pragma unroll
            for (int r = 0; r < 8; ++r) s[r] = 0.f;
            if (kk0 <= qend) {
#pragma unroll
                for (int c = 0; c < 4; ++c)
                    s = wmma_f16(qa[c], load_b_frag_kmajor(Kh, HD, kk0, c * 32), s);
            }
#pragma unroll
            for (int r = 0; r < 8; ++r) {
                int m = q0 + r + hi8;
                int n = kk0 + ncol;
                p[t][r] = (n > m || kk0 > qend) ? -1e30f : s[r] * scale;
            }
        }
#pragma unroll
        for (int r = 0; r < 8; ++r) {
            float tilemax = rowred_max(fmaxf(p[0][r], p[1][r]));
            float mx = fmaxf(rmax[r], tilemax);
            float alpha = __expf(rmax[r] - mx);
            rmax[r] = mx;
            float e0 = __expf(p[0][r] - mx);
            float e1 = __expf(p[1][r] - mx);
            rsum[r] = rsum[r] * alpha + rowred_sum(e0 + e1);
            p[0][r] = e0; p[1][r] = e1;
#pragma unroll
            for (int c = 0; c < 8; ++c) o[c][r] *= alpha;
        }
        // stage P (16x32 f16) through LDS to rebuild the A fragment layout
#pragma unroll
        for (int t = 0; t < 2; ++t)
#pragma unroll
            for (int r = 0; r < 8; ++r)
                pls[(r + hi8) * 32 + t * 16 + ncol] = (_Float16)p[t][r];
        __syncthreads();
        v16h pa = load_a_frag(pls, 32, 0, 0);
        __syncthreads();
        // o[:, c*16..c*16+15] += P(16x32) x V[k0..k0+31, c*16..]  via V^T
#pragma unroll
        for (int c = 0; c < 8; ++c)
            o[c] = wmma_f16(pa, load_b_frag_kmajor(VhT, SEQ, c * 16, k0), o[c]);
    }

#pragma unroll
    for (int r = 0; r < 8; ++r) {
        float inv = 1.0f / rsum[r];
        int m = q0 + r + hi8;
#pragma unroll
        for (int c = 0; c < 8; ++c)
            Om[(size_t)m * (NHQ * HD) + h * HD + c * 16 + ncol] = (_Float16)(o[c][r] * inv);
    }
}

// ---------------------------------------------------------------------------
// act = f16( silu(g) * u )
// ---------------------------------------------------------------------------
__global__ void __launch_bounds__(256) silu_mul_kernel(const float* __restrict__ g,
                                                       const float* __restrict__ u,
                                                       _Float16* __restrict__ act, int n) {
    int i = blockIdx.x * 256 + threadIdx.x;
    if (i < n) {
        float gv = g[i];
        float sg = gv / (1.0f + __expf(-gv));
        act[i] = (_Float16)(sg * u[i]);
    }
}

// ---------------------------------------------------------------------------
// Host-side orchestration
// ---------------------------------------------------------------------------
extern "C" void kernel_launch(void* const* d_in, const int* in_sizes, int n_in,
                              void* d_out, int out_size, void* d_ws, size_t ws_size,
                              hipStream_t stream) {
    const float* hs   = (const float*)d_in[0];
    const float* cosb = (const float*)d_in[1];
    const float* sinb = (const float*)d_in[2];
    // d_in[3] attention_mask: exactly causal, implemented in-kernel
    const float* Wq = (const float*)d_in[4];
    const float* Wk = (const float*)d_in[5];
    const float* Wv = (const float*)d_in[6];
    const float* Wo = (const float*)d_in[7];
    const float* Wg = (const float*)d_in[8];
    const float* Wu = (const float*)d_in[9];
    const float* Wd = (const float*)d_in[10];
    const float* ln1 = (const float*)d_in[11];
    const float* ln2 = (const float*)d_in[12];
    const float* qn  = (const float*)d_in[13];
    const float* kn  = (const float*)d_in[14];

    float* h = (float*)d_out;   // running residual stream
    hipMemcpyAsync(h, hs, (size_t)SEQ * DMODEL * sizeof(float),
                   hipMemcpyDeviceToDevice, stream);

    char* ws = (char*)d_ws;
    size_t off = 0;
    auto take = [&](size_t bytes) { char* p = ws + off; off += (bytes + 255) & ~size_t(255); return p; };
    _Float16* xf16  = (_Float16*)take((size_t)SEQ * DMODEL * 2);
    float*    qf32  = (float*)   take((size_t)SEQ * NHQ  * HD * 4);
    float*    kf32  = (float*)   take((size_t)SEQ * NHKV * HD * 4);
    float*    vf32  = (float*)   take((size_t)SEQ * NHKV * HD * 4);
    _Float16* qf16  = (_Float16*)take((size_t)NHQ  * SEQ * HD * 2);
    _Float16* kf16  = (_Float16*)take((size_t)NHKV * SEQ * HD * 2);
    _Float16* vTf16 = (_Float16*)take((size_t)NHKV * HD * SEQ * 2);
    _Float16* of16  = (_Float16*)take((size_t)SEQ * NHQ * HD * 2);
    float*    gf32  = (float*)   take((size_t)SEQ * DFF * 4);
    float*    uf32  = (float*)   take((size_t)SEQ * DFF * 4);
    _Float16* af16  = (_Float16*)take((size_t)SEQ * DFF * 2);
    (void)ws_size;

    const dim3 gemmBlk(32, 8);
    for (int l = 0; l < NLAYER; ++l) {
        const float* Wq_l = Wq + (size_t)l * DMODEL * (NHQ * HD);
        const float* Wk_l = Wk + (size_t)l * DMODEL * (NHKV * HD);
        const float* Wv_l = Wv + (size_t)l * DMODEL * (NHKV * HD);
        const float* Wo_l = Wo + (size_t)l * (NHQ * HD) * DMODEL;
        const float* Wg_l = Wg + (size_t)l * DMODEL * DFF;
        const float* Wu_l = Wu + (size_t)l * DMODEL * DFF;
        const float* Wd_l = Wd + (size_t)l * DFF * DMODEL;

        rmsnorm_f16_kernel<<<SEQ, 256, 0, stream>>>(h, ln1 + (size_t)l * DMODEL, xf16);

        gemm_f16w32_kernel<false><<<dim3((NHQ * HD) / 64, SEQ / 128), gemmBlk, 0, stream>>>(
            xf16, Wq_l, qf32, nullptr, SEQ, NHQ * HD, DMODEL);
        gemm_f16w32_kernel<false><<<dim3((NHKV * HD) / 64, SEQ / 128), gemmBlk, 0, stream>>>(
            xf16, Wk_l, kf32, nullptr, SEQ, NHKV * HD, DMODEL);
        gemm_f16w32_kernel<false><<<dim3((NHKV * HD) / 64, SEQ / 128), gemmBlk, 0, stream>>>(
            xf16, Wv_l, vf32, nullptr, SEQ, NHKV * HD, DMODEL);

        qk_rope_kernel<<<SEQ, 256, 0, stream>>>(qf32, kf32, vf32, cosb, sinb,
                                                qn + (size_t)l * HD, kn + (size_t)l * HD,
                                                qf16, kf16, vTf16);

        flash_attn_kernel<<<dim3(NHQ, SEQ / 16), 32, 0, stream>>>(qf16, kf16, vTf16, of16);

        gemm_f16w32_kernel<true><<<dim3(DMODEL / 64, SEQ / 128), gemmBlk, 0, stream>>>(
            of16, Wo_l, h, h, SEQ, DMODEL, NHQ * HD);

        rmsnorm_f16_kernel<<<SEQ, 256, 0, stream>>>(h, ln2 + (size_t)l * DMODEL, xf16);

        gemm_f16w32_kernel<false><<<dim3(DFF / 64, SEQ / 128), gemmBlk, 0, stream>>>(
            xf16, Wg_l, gf32, nullptr, SEQ, DFF, DMODEL);
        gemm_f16w32_kernel<false><<<dim3(DFF / 64, SEQ / 128), gemmBlk, 0, stream>>>(
            xf16, Wu_l, uf32, nullptr, SEQ, DFF, DMODEL);
        silu_mul_kernel<<<(SEQ * DFF + 255) / 256, 256, 0, stream>>>(gf32, uf32, af16, SEQ * DFF);
        gemm_f16w32_kernel<true><<<dim3(DMODEL / 64, SEQ / 128), gemmBlk, 0, stream>>>(
            af16, Wd_l, h, h, SEQ, DMODEL, DFF);
    }
}
